// attnHead_74955769250656
// MI455X (gfx1250) — matrix operand order repaired and checked by reading, not compiled, not run on hardware
//
#include <hip/hip_runtime.h>
#include <hip/hip_bf16.h>

typedef __attribute__((ext_vector_type(16))) _Float16 v16h;
typedef __attribute__((ext_vector_type(8)))  float    v8f;

union U16 {
    v16h v;
    unsigned int u[8];
    _Float16 h[16];
};

#define CTX   512
#define HID   256
#define HEADD 32
#define QK_STRIDE 34    // f16 per row (32 + pad), 17 dwords (odd) -> conflict-free
#define VT_STRIDE 514   // f16 per row (512 + pad), 257 dwords (odd)
#define W_STRIDE  258   // f16 per row (256 + pad), 129 dwords (odd)
#define ATTN_SCALE 0.17677669529663687f  // 32^-0.5

__device__ __forceinline__ int f32_order_key(float f) {
    int i = __float_as_int(f);
    return (i >= 0) ? i : (i ^ 0x7fffffff);
}
__device__ __forceinline__ float f32_order_unkey(int k) {
    return __int_as_float((k >= 0) ? k : (k ^ 0x7fffffff));
}

__global__ __launch_bounds__(256)
void attn_head_f16wmma_kernel(const float* __restrict__ E,
                              const float* __restrict__ Wk,
                              const float* __restrict__ Wq,
                              const float* __restrict__ Wv,
                              float* __restrict__ out) {
    // LDS: 2*512*34 + 32*514 + 3*32*258 + 8*16*34 f16  + 16*33 f32 + 128 B  = 163,008 B
    __shared__ _Float16 sQ[CTX * QK_STRIDE];
    __shared__ _Float16 sK[CTX * QK_STRIDE];
    __shared__ _Float16 sVt[HEADD * VT_STRIDE];          // transposed: [col][key]
    __shared__ _Float16 sW[3 * HEADD * W_STRIDE];        // [mat][n][k], mat: 0=Q,1=K,2=V
    __shared__ _Float16 sP[8 * 16 * QK_STRIDE];          // per-wave P staging [16][34]
    __shared__ float    sO[16 * 33];
    __shared__ int      sRowMaxI[16];
    __shared__ float    sRowSum[16];

    const int tid  = threadIdx.x;
    const int lane = tid & 31;
    const int wave = tid >> 5;
    const int hi   = lane >> 4;   // half-wave: 0 or 1
    const int nl   = lane & 15;

    const float* __restrict__ Eb = E + (size_t)blockIdx.x * (CTX * HID);

    // ---- Stage W (f16, column-major rows: sW[mat][n][k]) ----
    for (int i = tid; i < 3 * HID * HEADD; i += 256) {
        const int mat = i >> 13;         // / 8192
        const int rem = i & 8191;
        const int k   = rem >> 5;
        const int n   = rem & 31;
        const float* src = (mat == 0) ? Wq : (mat == 1) ? Wk : Wv;
        sW[mat * (HEADD * W_STRIDE) + n * W_STRIDE + k] = (_Float16)src[k * HEADD + n];
    }
    __syncthreads();

    // ---- Phase A: projections  Q,K,V = E @ W  (wmma f32.16x16x32.f16) ----
    // 3 matrices * 32 row-tiles * 2 col-tiles = 192 output tiles over 8 waves
    for (int t = wave; t < 192; t += 8) {
        const int mat = t >> 6;
        const int rt  = (t & 63) >> 1;
        const int ct  = t & 1;
        const int row = 16 * rt + nl;
        const float* __restrict__ erow = Eb + row * HID + 8 * hi;
        const unsigned int* __restrict__ wrow =
            (const unsigned int*)(sW + mat * (HEADD * W_STRIDE) + nl * W_STRIDE);

        v8f acc = {};
        #pragma unroll
        for (int ks = 0; ks < 8; ++ks) {
            // A-frag: lane holds E[row][32ks + 8hi + 0..7] and E[row][32ks + 16 + 8hi + 0..7]
            const float4 f0 = *(const float4*)(erow + 32 * ks + 0);
            const float4 f1 = *(const float4*)(erow + 32 * ks + 4);
            const float4 f2 = *(const float4*)(erow + 32 * ks + 16);
            const float4 f3 = *(const float4*)(erow + 32 * ks + 20);
            U16 A;
            A.h[0]  = (_Float16)f0.x; A.h[1]  = (_Float16)f0.y;
            A.h[2]  = (_Float16)f0.z; A.h[3]  = (_Float16)f0.w;
            A.h[4]  = (_Float16)f1.x; A.h[5]  = (_Float16)f1.y;
            A.h[6]  = (_Float16)f1.z; A.h[7]  = (_Float16)f1.w;
            A.h[8]  = (_Float16)f2.x; A.h[9]  = (_Float16)f2.y;
            A.h[10] = (_Float16)f2.z; A.h[11] = (_Float16)f2.w;
            A.h[12] = (_Float16)f3.x; A.h[13] = (_Float16)f3.y;
            A.h[14] = (_Float16)f3.z; A.h[15] = (_Float16)f3.w;
            // B-frag: dword d = packed pair W[32ks + 8hi + 2d .. +1][n] (d<4), W[32ks + 16 + 8hi + 2(d-4) .. +1][n] (d>=4)
            U16 B;
            #pragma unroll
            for (int d = 0; d < 8; ++d)
                B.u[d] = wrow[16 * ks + 4 * hi + d + 4 * (d >> 2)];
            acc = __builtin_amdgcn_wmma_f32_16x16x32_f16(
                false, A.v, false, B.v, (short)0, acc, false, false);
        }
        // store C (f32 C/D layout: lane -> m = v + 8*hi, n = nl)
        if (mat == 2) {
            _Float16* dst = sVt + (16 * ct + nl) * VT_STRIDE + 16 * rt + 8 * hi;
            #pragma unroll
            for (int v = 0; v < 8; ++v) dst[v] = (_Float16)acc[v];
        } else {
            _Float16* base = (mat == 0) ? sQ : sK;
            #pragma unroll
            for (int v = 0; v < 8; ++v)
                base[(16 * rt + v + 8 * hi) * QK_STRIDE + 16 * ct + nl] = (_Float16)acc[v];
        }
    }
    __syncthreads();

    // ---- Phase B: causal attention, one 16-row query tile at a time ----
    _Float16* __restrict__ Pw = sP + wave * (16 * QK_STRIDE);

    for (int qt = 0; qt < 32; ++qt) {
        for (int i = tid; i < 16 * 33; i += 256) sO[i] = 0.0f;
        if (tid < 16) { sRowMaxI[tid] = (int)0x80000000; sRowSum[tid] = 0.0f; }
        __syncthreads();

        // wave w owns key-tiles {w, w+8, w+16, w+24} intersected with [0, qt]
        const int cnt = (qt >= wave) ? (((qt - wave) >> 3) + 1) : 0;

        // Q A-frag for this row tile (rows m = 16qt + nl; K runs 8hi+[0..7], 16+8hi+[0..7])
        U16 Aq;
        {
            const char* qrow = (const char*)sQ + (16 * qt + nl) * (2 * QK_STRIDE) + 16 * hi;
            #pragma unroll
            for (int d = 0; d < 4; ++d) {
                Aq.u[d]     = *(const unsigned int*)(qrow + 4 * d);
                Aq.u[4 + d] = *(const unsigned int*)(qrow + 32 + 4 * d);
            }
        }

        float S[4][8];
        float rm[8];
        #pragma unroll
        for (int v = 0; v < 8; ++v) rm[v] = -__builtin_inff();

        #pragma unroll
        for (int j = 0; j < 4; ++j) {
            if (j < cnt) {                       // wave-uniform guard: EXEC all-ones for WMMA
                const int kt = wave + 8 * j;
                U16 Bk;                           // B = K^T: K runs 8hi+[0..7], 16+8hi+[0..7] (same as Q frag)
                const char* krow = (const char*)sK + (16 * kt + nl) * (2 * QK_STRIDE) + 16 * hi;
                #pragma unroll
                for (int d = 0; d < 8; ++d)
                    Bk.u[d] = *(const unsigned int*)(krow + 4 * d + 16 * (d >> 2));
                v8f cz = {};
                v8f dres = __builtin_amdgcn_wmma_f32_16x16x32_f16(
                    false, Aq.v, false, Bk.v, (short)0, cz, false, false);
                #pragma unroll
                for (int v = 0; v < 8; ++v) {
                    float s = dres[v] * ATTN_SCALE;
                    if (kt == qt && nl > v + 8 * hi) s = -__builtin_inff();  // causal diag
                    S[j][v] = s;
                    rm[v] = fmaxf(rm[v], s);
                }
            }
        }

        // row-max: butterfly over the 16-lane half, then LDS atomic (int-keyed ds_max_i32)
        #pragma unroll
        for (int v = 0; v < 8; ++v) {
            float m = rm[v];
            m = fmaxf(m, __shfl_xor(m, 1, 16));
            m = fmaxf(m, __shfl_xor(m, 2, 16));
            m = fmaxf(m, __shfl_xor(m, 4, 16));
            m = fmaxf(m, __shfl_xor(m, 8, 16));
            rm[v] = m;
        }
        if (cnt > 0 && (lane & 15) == 0) {
            #pragma unroll
            for (int v = 0; v < 8; ++v)
                atomicMax(&sRowMaxI[v + 8 * hi], f32_order_key(rm[v]));
        }
        __syncthreads();

        float rmaxv[8];
        #pragma unroll
        for (int v = 0; v < 8; ++v) rmaxv[v] = f32_order_unkey(sRowMaxI[v + 8 * hi]);

        float lsum[8] = {};
        v8f O0 = {}, O1 = {};

        #pragma unroll
        for (int jp = 0; jp < 4; jp += 2) {
            if (jp < cnt) {                      // wave-uniform
                const int ka   = wave + 8 * jp;
                const bool hasB = (jp + 1) < cnt;
                const int kb   = hasB ? (wave + 8 * (jp + 1)) : ka;
                // stage unnormalized P (f16) transposed into A-layout rows
                #pragma unroll
                for (int v = 0; v < 8; ++v) {
                    float pa = __expf(S[jp][v] - rmaxv[v]);
                    lsum[v] += pa;
                    Pw[(v + 8 * hi) * QK_STRIDE + nl] = (_Float16)pa;
                    float pb = 0.0f;
                    if (hasB) { pb = __expf(S[jp + 1][v] - rmaxv[v]); lsum[v] += pb; }
                    Pw[(v + 8 * hi) * QK_STRIDE + 16 + nl] = (_Float16)pb;
                }
                U16 Ap;                           // P A-frag (contiguous: K runs 16hi+[0..15])
                const char* prow = (const char*)Pw + nl * (2 * QK_STRIDE) + 32 * hi;
                #pragma unroll
                for (int d = 0; d < 4; ++d) {
                    Ap.u[d]     = *(const unsigned int*)(prow + 4 * d);
                    Ap.u[4 + d] = *(const unsigned int*)(prow + 16 + 4 * d);
                }
                #pragma unroll
                for (int c = 0; c < 2; ++c) {     // B = V[keys of (ka,kb)][16c + nl]
                    U16 Bv;
                    const char* vrow = (const char*)sVt + (16 * c + nl) * (2 * VT_STRIDE)
                                     + 32 * (hi ? kb : ka);
                    #pragma unroll
                    for (int d = 0; d < 8; ++d)
                        Bv.u[d] = *(const unsigned int*)(vrow + 4 * d);
                    if (c == 0)
                        O0 = __builtin_amdgcn_wmma_f32_16x16x32_f16(
                            false, Ap.v, false, Bv.v, (short)0, O0, false, false);
                    else
                        O1 = __builtin_amdgcn_wmma_f32_16x16x32_f16(
                            false, Ap.v, false, Bv.v, (short)0, O1, false, false);
                }
            }
        }

        // row-sum butterfly + LDS ds_add_f32; O cross-wave reduce via ds_add_f32
        #pragma unroll
        for (int v = 0; v < 8; ++v) {
            float s = lsum[v];
            s += __shfl_xor(s, 1, 16);
            s += __shfl_xor(s, 2, 16);
            s += __shfl_xor(s, 4, 16);
            s += __shfl_xor(s, 8, 16);
            lsum[v] = s;
        }
        if (cnt > 0) {
            if ((lane & 15) == 0) {
                #pragma unroll
                for (int v = 0; v < 8; ++v)
                    atomicAdd(&sRowSum[v + 8 * hi], lsum[v]);
            }
            #pragma unroll
            for (int v = 0; v < 8; ++v) {
                atomicAdd(&sO[(v + 8 * hi) * 33 + nl],      O0[v]);
                atomicAdd(&sO[(v + 8 * hi) * 33 + 16 + nl], O1[v]);
            }
        }
        __syncthreads();

        // normalize + store this query tile
        for (int i = tid; i < 16 * HEADD; i += 256) {
            const int m = i >> 5;
            const int n = i & 31;
            out[((size_t)blockIdx.x * CTX + 16 * qt + m) * HEADD + n] = sO[m * 33 + n] / sRowSum[m];
        }
        __syncthreads();
    }
}

extern "C" void kernel_launch(void* const* d_in, const int* in_sizes, int n_in,
                              void* d_out, int out_size, void* d_ws, size_t ws_size,
                              hipStream_t stream) {
    const float* E  = (const float*)d_in[0];   // [512, 512, 256]
    const float* Wk = (const float*)d_in[1];   // [256, 32]
    const float* Wq = (const float*)d_in[2];   // [256, 32]
    const float* Wv = (const float*)d_in[3];   // [256, 32]
    float* out = (float*)d_out;                // [512, 512, 32]
    (void)in_sizes; (void)n_in; (void)out_size; (void)d_ws; (void)ws_size;
    attn_head_f16wmma_kernel<<<512, 256, 0, stream>>>(E, Wk, Wq, Wv, out);
}
